// ElecElecEdges_32418413150493
// MI455X (gfx1250) — matrix operand order repaired and checked by reading, not compiled, not run on hardware
//
#include <hip/hip_runtime.h>
#include <hip/hip_bf16.h>

typedef _Float16 v16h __attribute__((ext_vector_type(16)));
typedef _Float16 h8   __attribute__((ext_vector_type(8)));
typedef float    v8f  __attribute__((ext_vector_type(8)));

#define BN_PAIRS (32768 * 32)
#define HALF_OUT (BN_PAIRS * 32)

// A-fragment (16x32 f16, MxK): lane holds 16 contiguous halves at
// K = k0 + hi*16 + [0..15]
__device__ __forceinline__ v16h ld_a_frag(const _Float16* p) {
  h8 lo = *(const h8*)(p);
  h8 hi = *(const h8*)(p + 8);
  return __builtin_shufflevector(lo, hi, 0,1,2,3,4,5,6,7,8,9,10,11,12,13,14,15);
}
// B-fragment (32x16 f16, KxN) from K-contiguous (transposed) weight storage:
// lane reads 16 contiguous halves K = k0 + hi*16 + [0..15] at column N = lane&15
__device__ __forceinline__ v16h ld_b_frag(const _Float16* p) {
  h8 lo = *(const h8*)(p);
  h8 hi = *(const h8*)(p + 8);
  return __builtin_shufflevector(lo, hi, 0,1,2,3,4,5,6,7,8,9,10,11,12,13,14,15);
}

__global__ __launch_bounds__(128)
void ElecElecEdges_kernel(
    const float* __restrict__ r,      // [B,3]
    const float* __restrict__ r_nb,   // [B,N,3]
    const int*   __restrict__ s,      // [B]
    const int*   __restrict__ s_nb,   // [B,N]
    const float* __restrict__ scales, // [16]
    const float* __restrict__ kern,   // [4,64]
    const float* __restrict__ bias,   // [64]
    const float* __restrict__ w_beta, // [80,64]
    const float* __restrict__ w_same, // [64,64]
    const float* __restrict__ w_diff, // [64,64]
    float* __restrict__ out)          // [2 * B*N * 32]
{
  // f16 tiles; strides padded (104 / 72 halves) for 16B alignment + bank spread
  __shared__ __attribute__((aligned(16))) _Float16 Xlds[64 * 104];   // [row][K<=96]
  __shared__ __attribute__((aligned(16))) _Float16 betaLds[64 * 72]; // [row][K=64]
  __shared__ __attribute__((aligned(16))) _Float16 wbetaT[64 * 104]; // [N][K<=96]
  __shared__ __attribute__((aligned(16))) _Float16 wsameT[64 * 72];  // [N][K=64]
  __shared__ __attribute__((aligned(16))) _Float16 wdiffT[64 * 72];  // [N][K=64]
  __shared__ float kernelLds[256];
  __shared__ float biasLds[64];
  __shared__ float invs2[16];
  __shared__ float featLds[64 * 4];
  __shared__ float cutLds[64];
  __shared__ int   flagLds[64];

  const int tid = threadIdx.x;
  const int blockBase = blockIdx.x * 64;   // 64 pairs per block

  // ---- Phase 0: stage weights into LDS (transpose + f16, zero-pad K 80->96)
  for (int i = tid; i < 64 * 96; i += 128) {
    int n = i / 96, k = i - n * 96;
    wbetaT[n * 104 + k] = (k < 80) ? (_Float16)w_beta[k * 64 + n] : (_Float16)0.0f;
  }
  for (int i = tid; i < 64 * 64; i += 128) {
    int n = i >> 6, k = i & 63;
    wsameT[n * 72 + k] = (_Float16)w_same[k * 64 + n];
    wdiffT[n * 72 + k] = (_Float16)w_diff[k * 64 + n];
  }
  for (int i = tid; i < 256; i += 128) kernelLds[i] = kern[i];
  if (tid < 64) biasLds[tid] = bias[tid];
  if (tid < 16) { float sc = scales[tid]; invs2[tid] = 1.0f / (sc * sc); }

  // ---- Phase 1: per-pair diff features, cutoff, spin flag
  if (tid < 64) {
    int p = blockBase + tid;          // flat pair index b*32 + n
    int b = p >> 5;
    float dx = r[b * 3 + 0] - r_nb[p * 3 + 0];
    float dy = r[b * 3 + 1] - r_nb[p * 3 + 1];
    float dz = r[b * 3 + 2] - r_nb[p * 3 + 2];
    float d  = sqrtf(dx * dx + dy * dy + dz * dz);
    featLds[tid * 4 + 0] = dx;
    featLds[tid * 4 + 1] = dy;
    featLds[tid * 4 + 2] = dz;
    featLds[tid * 4 + 3] = d;
    // 0.5*(cos(pi*d/5)+1) for d<5 else 0
    cutLds[tid]  = (d < 5.0f) ? 0.5f * (cosf(0.62831853071795864769f * d) + 1.0f) : 0.0f;
    flagLds[tid] = (s[b] == s_nb[p]) ? 1 : 0;
  }
  __syncthreads();

  // ---- Phase 1b: X[64][96] = [tanh(feat@kernel+bias) | env | 0-pad]  (f16)
  for (int i = tid; i < 64 * 96; i += 128) {
    int row = i / 96, c = i - row * 96;
    float val;
    if (c < 64) {
      float acc = biasLds[c];
      acc += featLds[row * 4 + 0] * kernelLds[c];
      acc += featLds[row * 4 + 1] * kernelLds[64 + c];
      acc += featLds[row * 4 + 2] * kernelLds[128 + c];
      acc += featLds[row * 4 + 3] * kernelLds[192 + c];
      val = tanhf(acc);
    } else if (c < 80) {
      float d = featLds[row * 4 + 3];
      val = expf(-d * d * invs2[c - 64]);
    } else {
      val = 0.0f;
    }
    Xlds[row * 104 + c] = (_Float16)val;
  }
  __syncthreads();

  // ---- Phase 2: WMMA GEMMs. Each wave owns 16 rows.
  const int wave = tid >> 5, lane = tid & 31;
  const int hi = lane >> 4, nl = lane & 15;
  const int rowA = wave * 16 + nl;   // A-fragment row for this lane

  // beta[16x64] = X[16x96] @ w_beta[96x64], scaled by cut, stored f16 in LDS
  #pragma unroll
  for (int nt = 0; nt < 4; ++nt) {
    const int n0 = nt * 16;
    v8f acc = {};
    const _Float16* xrow = &Xlds[rowA * 104 + hi * 16];
    const _Float16* wb   = &wbetaT[(n0 + nl) * 104 + hi * 16];
    #pragma unroll
    for (int kk = 0; kk < 3; ++kk) {
      v16h a  = ld_a_frag(xrow + kk * 32);
      v16h bm = ld_b_frag(wb + kk * 32);
      acc = __builtin_amdgcn_wmma_f32_16x16x32_f16(false, a, false, bm,
                                                   (short)0, acc, false, false);
    }
    #pragma unroll
    for (int v = 0; v < 8; ++v) {
      int row = wave * 16 + hi * 8 + v;             // C/D: M = v + 8*hi
      betaLds[row * 72 + n0 + nl] = (_Float16)(acc[v] * cutLds[row]);
    }
  }
  __syncthreads();

  // gamma_same / gamma_diff = beta[16x64] @ w[64x64]; spin-select; store
  #pragma unroll
  for (int nt = 0; nt < 4; ++nt) {
    const int n0 = nt * 16;
    v8f accS = {}, accD = {};
    const _Float16* brow = &betaLds[rowA * 72 + hi * 16];
    const _Float16* ws   = &wsameT[(n0 + nl) * 72 + hi * 16];
    const _Float16* wd   = &wdiffT[(n0 + nl) * 72 + hi * 16];
    #pragma unroll
    for (int kk = 0; kk < 2; ++kk) {
      v16h a  = ld_a_frag(brow + kk * 32);
      v16h bs = ld_b_frag(ws + kk * 32);
      v16h bd = ld_b_frag(wd + kk * 32);
      accS = __builtin_amdgcn_wmma_f32_16x16x32_f16(false, a, false, bs,
                                                    (short)0, accS, false, false);
      accD = __builtin_amdgcn_wmma_f32_16x16x32_f16(false, a, false, bd,
                                                    (short)0, accD, false, false);
    }
    const int c = n0 + nl;                         // gamma column 0..63
    const int cbase = (c < 32) ? c : (HALF_OUT + (c - 32));
    #pragma unroll
    for (int v = 0; v < 8; ++v) {
      int row = wave * 16 + hi * 8 + v;
      int p = blockBase + row;
      float g = flagLds[row] ? accS[v] : accD[v];
      out[cbase + p * 32] = g;                     // coalesced across lanes
    }
  }
}

extern "C" void kernel_launch(void* const* d_in, const int* in_sizes, int n_in,
                              void* d_out, int out_size, void* d_ws, size_t ws_size,
                              hipStream_t stream) {
  const float* r      = (const float*)d_in[0];
  const float* r_nb   = (const float*)d_in[1];
  const int*   s      = (const int*)  d_in[2];
  const int*   s_nb   = (const int*)  d_in[3];
  const float* scales = (const float*)d_in[4];
  const float* kern   = (const float*)d_in[5];
  const float* bias   = (const float*)d_in[6];
  const float* w_beta = (const float*)d_in[7];
  const float* w_same = (const float*)d_in[8];
  const float* w_diff = (const float*)d_in[9];
  float* out = (float*)d_out;

  const int nblocks = BN_PAIRS / 64;   // 16384 blocks, 64 pairs each
  ElecElecEdges_kernel<<<nblocks, 128, 0, stream>>>(
      r, r_nb, s, s_nb, scales, kern, bias, w_beta, w_same, w_diff, out);
}
